// MultiheadAttention_43138651521273
// MI455X (gfx1250) — compile-verified, matrix operand order checked by reading, not run on hardware
//
#include <hip/hip_runtime.h>
#include <hip/hip_bf16.h>

// ---------------------------------------------------------------------------
// Multi-head attention for MI455X (gfx1250), bf16 WMMA pipeline.
//   - projections: 128x128-tile GEMM, K staged 64-wide, double-buffered
//     async global->LDS copies (ASYNCcnt) overlapped with WMMA
//   - V projection writes V^T per head so flash P@V B-fragments are
//     contiguous ds_load_b128 (no strided LDS gathers)
//   - flash attention: 128 q-rows/block, 64-key tiles, online softmax with
//     DPP16 ROW_XMASK butterflies (pure VALU)
// ---------------------------------------------------------------------------

typedef __bf16 bf16_t;
typedef __attribute__((ext_vector_type(16))) __bf16 v16bf;
typedef __attribute__((ext_vector_type(8)))  __bf16 v8bf;
typedef __attribute__((ext_vector_type(8)))  float  v8f;

#define EMB    1024
#define NHEAD  16
#define HD     64
#define BATCH  4
#define SEQ    2048
#define TOKENS (BATCH * SEQ)

// ---------------------------------------------------------------------------
// CDNA5 async global->LDS copy (tracked by ASYNCcnt)
// ---------------------------------------------------------------------------
#if defined(__has_builtin)
#if __has_builtin(__builtin_amdgcn_global_load_async_to_lds_b128)
#define HAVE_ASYNC_COPY 1
#endif
#endif
#ifndef HAVE_ASYNC_COPY
#define HAVE_ASYNC_COPY 0
#endif

#if HAVE_ASYNC_COPY
typedef int v4i_vs __attribute__((vector_size(16)));
typedef __attribute__((address_space(1))) v4i_vs* as1_v4i_ptr;
typedef __attribute__((address_space(3))) v4i_vs* as3_v4i_ptr;
#endif

__device__ __forceinline__ void copy16_to_lds(bf16_t* lds, const bf16_t* g) {
#if HAVE_ASYNC_COPY
    __builtin_amdgcn_global_load_async_to_lds_b128(
        (as1_v4i_ptr)(void*)const_cast<bf16_t*>(g),
        (as3_v4i_ptr)(void*)lds, 0, 0);
#else
    *(v8bf*)lds = *(const v8bf*)g;
#endif
}

__device__ __forceinline__ void async_copy_drain() {
#if HAVE_ASYNC_COPY
    asm volatile("s_wait_asynccnt 0" ::: "memory");
#endif
}

// ---------------------------------------------------------------------------
// WMMA helpers
// ---------------------------------------------------------------------------
__device__ __forceinline__ v8f wmma_bf16(v16bf a, v16bf b, v8f c) {
    return __builtin_amdgcn_wmma_f32_16x16x32_bf16(
        false, a, false, b, (short)0, c, false, false);
}

// A-fragment (and B-fragment for N x K row-major sources).
// lane -> row = lane&15; element e -> k = (e>=8?16:0) + (lane>=16?8:0) + (e&7)
// => two contiguous 16B loads per lane.
__device__ __forceinline__ v16bf load_frag_nk(const bf16_t* base, int ld) {
    const int lane = threadIdx.x & 31;
    const int row  = lane & 15;
    const int khi  = (lane & 16) >> 1;
    const bf16_t* p = base + row * ld + khi;
    v8bf lo = *(const v8bf*)(p);
    v8bf hi = *(const v8bf*)(p + 16);
    return __builtin_shufflevector(lo, hi,
        0, 1, 2, 3, 4, 5, 6, 7, 8, 9, 10, 11, 12, 13, 14, 15);
}

// ---------------------------------------------------------------------------
// DPP16 ROW_XMASK xor-butterfly (pure VALU; stays inside each 16-lane row)
// ---------------------------------------------------------------------------
template <int MASK>
__device__ __forceinline__ float dpp_xmask(float x) {
    int xi = __float_as_int(x);
    int r = __builtin_amdgcn_update_dpp(xi, xi, 0x160 | MASK, 0xF, 0xF, true);
    return __int_as_float(r);
}

__device__ __forceinline__ float row_max16(float x) {
    x = fmaxf(x, dpp_xmask<1>(x));
    x = fmaxf(x, dpp_xmask<2>(x));
    x = fmaxf(x, dpp_xmask<4>(x));
    x = fmaxf(x, dpp_xmask<8>(x));
    return x;
}

__device__ __forceinline__ float row_sum16(float x) {
    x += dpp_xmask<1>(x);
    x += dpp_xmask<2>(x);
    x += dpp_xmask<4>(x);
    x += dpp_xmask<8>(x);
    return x;
}

// ---------------------------------------------------------------------------
// f32 -> bf16 conversion (one-time)
// ---------------------------------------------------------------------------
__global__ __launch_bounds__(256) void cvt_f32_bf16(
    const float* __restrict__ src, bf16_t* __restrict__ dst, int n) {
    int i = (blockIdx.x * 256 + threadIdx.x) * 8;
    if (i + 8 <= n) {
        float4 f0 = *(const float4*)(src + i);
        float4 f1 = *(const float4*)(src + i + 4);
        v8bf o;
        o[0] = (bf16_t)f0.x; o[1] = (bf16_t)f0.y;
        o[2] = (bf16_t)f0.z; o[3] = (bf16_t)f0.w;
        o[4] = (bf16_t)f1.x; o[5] = (bf16_t)f1.y;
        o[6] = (bf16_t)f1.z; o[7] = (bf16_t)f1.w;
        *(v8bf*)(dst + i) = o;
    }
}

// ---------------------------------------------------------------------------
// GEMM:  Y = (X @ W^T + bias) * scale
// Block tile 128 tokens x 128 channels; K staged 64-wide, double-buffered.
// MODE 0: Y f32 row-major [tok][EMB]
// MODE 1: Y bf16 per-head [b][h][l][hd]        (Q, K)
// MODE 2: Y bf16 per-head transposed [b][h][hd][l]  (V -> contiguous stores)
// ---------------------------------------------------------------------------
template <int MODE, typename OutT>
__global__ __launch_bounds__(256) void gemm_xwT(
    const bf16_t* __restrict__ X,
    const bf16_t* __restrict__ W,
    const float*  __restrict__ bias,
    OutT* __restrict__ Y,
    float scale) {
    __shared__ __align__(16) bf16_t Xs[2][128 * 64];
    __shared__ __align__(16) bf16_t Ws[2][128 * 64];

    const int tid  = threadIdx.x;
    const int wave = tid >> 5;
    const int lane = tid & 31;
    const int tok0 = (blockIdx.x >> 3) * 128;
    const int n0b  = (blockIdx.x & 7) * 128;
    const int mw   = wave >> 1;   // 0..3
    const int nw   = wave & 1;    // 0..1

    auto stage = [&](int k0, int buf) {
#pragma unroll
        for (int t = 0; t < 4; ++t) {
            int idx = tid * 8 + t * 2048;
            int row = idx >> 6, col = idx & 63;
            copy16_to_lds(&Xs[buf][idx], X + (size_t)(tok0 + row) * EMB + k0 + col);
            copy16_to_lds(&Ws[buf][idx], W + (size_t)(n0b + row) * EMB + k0 + col);
        }
    };

    v8f acc[2][4];
#pragma unroll
    for (int i = 0; i < 2; ++i)
#pragma unroll
        for (int j = 0; j < 4; ++j) acc[i][j] = (v8f){};

    stage(0, 0);                    // prologue: tile 0 in flight
    int buf = 0;
    for (int k0 = 0; k0 < EMB; k0 += 64) {
        async_copy_drain();         // tile k0 landed (this wave's share)
        __syncthreads();            // all waves' shares landed
        if (k0 + 64 < EMB) stage(k0 + 64, buf ^ 1);   // overlap next tile

#pragma unroll
        for (int ks = 0; ks < 64; ks += 32) {
            v16bf a0 = load_frag_nk(&Xs[buf][(mw * 2 + 0) * 16 * 64 + ks], 64);
            v16bf a1 = load_frag_nk(&Xs[buf][(mw * 2 + 1) * 16 * 64 + ks], 64);
#pragma unroll
            for (int j = 0; j < 4; ++j) {
                v16bf b = load_frag_nk(&Ws[buf][(nw * 4 + j) * 16 * 64 + ks], 64);
                acc[0][j] = wmma_bf16(a0, b, acc[0][j]);
                acc[1][j] = wmma_bf16(a1, b, acc[1][j]);
            }
        }
        buf ^= 1;
    }

    // Epilogue. D layout: lane -> col n = lane&15; vgpr v -> row v + 8*(lane>=16)
    const int colL = lane & 15;
    const int rhi  = (lane >> 4) << 3;
#pragma unroll
    for (int i = 0; i < 2; ++i) {
#pragma unroll
        for (int j = 0; j < 4; ++j) {
            int nn = n0b + (nw * 4 + j) * 16 + colL;
            float bv = bias[nn];
            int tokb = tok0 + (mw * 2 + i) * 16 + rhi;  // 8 consecutive tokens
            if (MODE == 2) {
                // V^T: one contiguous 16B store of 8 consecutive l positions.
                int h = nn >> 6, d = nn & 63;
                int b = tokb / SEQ, l0 = tokb % SEQ;
                v8bf pk;
#pragma unroll
                for (int v = 0; v < 8; ++v)
                    pk[v] = (bf16_t)((acc[i][j][v] + bv) * scale);
                *(v8bf*)((bf16_t*)Y +
                         (((size_t)b * NHEAD + h) * HD + d) * SEQ + l0) = pk;
            } else {
#pragma unroll
                for (int v = 0; v < 8; ++v) {
                    int tok = tokb + v;
                    float val = (acc[i][j][v] + bv) * scale;
                    if (MODE == 1) {
                        int b = tok / SEQ, l = tok % SEQ;
                        int h = nn >> 6, d = nn & 63;
                        ((bf16_t*)Y)[(((size_t)b * NHEAD + h) * SEQ + l) * HD + d] =
                            (bf16_t)val;
                    } else {
                        Y[(size_t)tok * EMB + nn] = (OutT)val;
                    }
                }
            }
        }
    }
}

// ---------------------------------------------------------------------------
// Flash attention. One block = 128 q rows of one (b,h); 8 waves x 16 q rows.
// 64-key K / V^T tiles double-buffered in LDS. Q pre-scaled by 1/8.
// ---------------------------------------------------------------------------
__global__ __launch_bounds__(256) void flash_attn(
    const bf16_t* __restrict__ Q,    // [B*H][SEQ][HD]
    const bf16_t* __restrict__ K,    // [B*H][SEQ][HD]
    const bf16_t* __restrict__ Vt,   // [B*H][HD][SEQ]
    bf16_t* __restrict__ ctx) {      // [B][SEQ][EMB]
    __shared__ __align__(16) bf16_t Ks[2][64 * 64];
    __shared__ __align__(16) bf16_t Vs[2][64 * 64];   // V^T tile: [d][key]
    __shared__ __align__(16) bf16_t Ps[8][16 * 64];

    const int tid  = threadIdx.x;
    const int wave = tid >> 5;
    const int lane = tid & 31;
    const int bh   = blockIdx.x >> 4;
    const int qt   = blockIdx.x & 15;
    const size_t baseK = (size_t)bh * SEQ * HD;
    const size_t baseV = (size_t)bh * HD * SEQ;
    const int qrow0 = qt * 128 + wave * 16;

    const bf16_t* qptr = Q + baseK + (size_t)qrow0 * HD;
    v16bf qa0 = load_frag_nk(qptr, HD);        // d = 0..31
    v16bf qa1 = load_frag_nk(qptr + 32, HD);   // d = 32..63

    auto stage = [&](int kt, int buf) {
#pragma unroll
        for (int t = 0; t < 2; ++t) {
            int idx = tid * 8 + t * 2048;
            int r = idx >> 6, c = idx & 63;
            // K tile rows are keys; V^T tile rows are head dims.
            copy16_to_lds(&Ks[buf][idx], K  + baseK + (size_t)(kt + r) * HD  + c);
            copy16_to_lds(&Vs[buf][idx], Vt + baseV + (size_t)r * SEQ + kt + c);
        }
    };

    v8f o[4];
#pragma unroll
    for (int n = 0; n < 4; ++n) o[n] = (v8f){};
    float mrow[8], lrow[8];
#pragma unroll
    for (int v = 0; v < 8; ++v) { mrow[v] = -1e30f; lrow[v] = 0.0f; }

    const int colL = lane & 15;
    const int rhi  = (lane >> 4) << 3;

    stage(0, 0);                    // prologue: first K/V tile in flight
    int buf = 0;
    for (int kt = 0; kt < SEQ; kt += 64) {
        async_copy_drain();
        __syncthreads();
        if (kt + 64 < SEQ) stage(kt + 64, buf ^ 1);

        // Scores: 4 key-subtiles of 16; each accumulates over d = 64 (2 WMMA).
        v8f s[4];
#pragma unroll
        for (int sub = 0; sub < 4; ++sub) {
            s[sub] = (v8f){};
            s[sub] = wmma_bf16(qa0, load_frag_nk(&Ks[buf][sub * 16 * 64], 64), s[sub]);
            s[sub] = wmma_bf16(qa1, load_frag_nk(&Ks[buf][sub * 16 * 64 + 32], 64), s[sub]);
        }

        // Online softmax; row m = v + rhi lives in one 16-lane DPP row.
        float alpha[8];
        bf16_t* pw = Ps[wave];
#pragma unroll
        for (int v = 0; v < 8; ++v) {
            float mx = fmaxf(fmaxf(s[0][v], s[1][v]), fmaxf(s[2][v], s[3][v]));
            mx = row_max16(mx);
            float mnew = fmaxf(mrow[v], mx);
            alpha[v] = __expf(mrow[v] - mnew);
            mrow[v] = mnew;
            float rs = 0.0f;
#pragma unroll
            for (int sub = 0; sub < 4; ++sub) {
                float p = __expf(s[sub][v] - mnew);
                s[sub][v] = p;
                rs += p;
            }
            rs = row_sum16(rs);
            lrow[v] = lrow[v] * alpha[v] + rs;
            // D-layout -> A-layout: park P row in wave-private LDS.
            int m = v + rhi;
#pragma unroll
            for (int sub = 0; sub < 4; ++sub)
                pw[m * 64 + sub * 16 + colL] = (bf16_t)s[sub][v];
        }
        // Wave-local LDS ordering before the fragment re-load.
        asm volatile("s_wait_dscnt 0" ::: "memory");
        v16bf pa0 = load_frag_nk(pw, 64);        // keys 0..31
        v16bf pa1 = load_frag_nk(pw + 32, 64);   // keys 32..63

        // O = O*alpha + P(16x64) @ V(64x64); B-fragment from V^T is two b128s.
#pragma unroll
        for (int n = 0; n < 4; ++n) {
#pragma unroll
            for (int v = 0; v < 8; ++v) o[n][v] *= alpha[v];
            o[n] = wmma_bf16(pa0, load_frag_nk(&Vs[buf][n * 16 * 64], 64), o[n]);
            o[n] = wmma_bf16(pa1, load_frag_nk(&Vs[buf][n * 16 * 64 + 32], 64), o[n]);
        }
        buf ^= 1;
    }

    // Normalize and scatter to [b][l][h*64+d]
    const int b = bh >> 4, h = bh & 15;
#pragma unroll
    for (int n = 0; n < 4; ++n) {
        int d = n * 16 + colL;
#pragma unroll
        for (int v = 0; v < 8; ++v) {
            int l = qrow0 + v + rhi;
            float val = o[n][v] / lrow[v];
            ctx[((size_t)b * SEQ + l) * EMB + h * HD + d] = (bf16_t)val;
        }
    }
}

// ---------------------------------------------------------------------------
// Host launch
// ---------------------------------------------------------------------------
extern "C" void kernel_launch(void* const* d_in, const int* in_sizes, int n_in,
                              void* d_out, int out_size, void* d_ws, size_t ws_size,
                              hipStream_t stream) {
    const float* x  = (const float*)d_in[0];
    const float* Wq = (const float*)d_in[1];
    const float* bq = (const float*)d_in[2];
    const float* Wk = (const float*)d_in[3];
    const float* bk = (const float*)d_in[4];
    const float* Wv = (const float*)d_in[5];
    const float* bv = (const float*)d_in[6];
    const float* Wo = (const float*)d_in[7];
    const float* bo = (const float*)d_in[8];
    float* out = (float*)d_out;

    bf16_t* xb  = (bf16_t*)d_ws;                       // TOKENS*EMB
    bf16_t* wqb = xb  + (size_t)TOKENS * EMB;
    bf16_t* wkb = wqb + (size_t)EMB * EMB;
    bf16_t* wvb = wkb + (size_t)EMB * EMB;
    bf16_t* wob = wvb + (size_t)EMB * EMB;
    bf16_t* Qb  = wob + (size_t)EMB * EMB;             // [B*H][SEQ][HD]
    bf16_t* Kb  = Qb  + (size_t)TOKENS * EMB;
    bf16_t* Vtb = Kb  + (size_t)TOKENS * EMB;          // [B*H][HD][SEQ]
    bf16_t* Cb  = Vtb + (size_t)TOKENS * EMB;          // ctx [B][SEQ][EMB]

    const int NX = TOKENS * EMB;
    const int NW = EMB * EMB;
    cvt_f32_bf16<<<NX / 2048, 256, 0, stream>>>(x,  xb,  NX);
    cvt_f32_bf16<<<NW / 2048, 256, 0, stream>>>(Wq, wqb, NW);
    cvt_f32_bf16<<<NW / 2048, 256, 0, stream>>>(Wk, wkb, NW);
    cvt_f32_bf16<<<NW / 2048, 256, 0, stream>>>(Wv, wvb, NW);
    cvt_f32_bf16<<<NW / 2048, 256, 0, stream>>>(Wo, wob, NW);

    const int GEMM_GRID = (TOKENS / 128) * (EMB / 128);  // 512
    const float qscale = 0.125f;                         // 1/sqrt(64)
    gemm_xwT<1, bf16_t><<<GEMM_GRID, 256, 0, stream>>>(xb, wqb, bq, Qb,  qscale);
    gemm_xwT<1, bf16_t><<<GEMM_GRID, 256, 0, stream>>>(xb, wkb, bk, Kb,  1.0f);
    gemm_xwT<2, bf16_t><<<GEMM_GRID, 256, 0, stream>>>(xb, wvb, bv, Vtb, 1.0f);

    flash_attn<<<BATCH * NHEAD * (SEQ / 128), 256, 0, stream>>>(Qb, Kb, Vtb, Cb);

    gemm_xwT<0, float><<<GEMM_GRID, 256, 0, stream>>>(Cb, wob, bo, out, 1.0f);
}
